// RichSemanticEncoder_14018773254544
// MI455X (gfx1250) — matrix operand order repaired and checked by reading, not compiled, not run on hardware
//
#include <hip/hip_runtime.h>
#include <hip/hip_bf16.h>

typedef __attribute__((ext_vector_type(16))) __bf16 v16bf;
typedef __attribute__((ext_vector_type(8)))  __bf16 v8bf;
typedef __attribute__((ext_vector_type(8)))  float  v8f;

#define NROWS 32768
#define LNEPS 1e-5f

__device__ __forceinline__ unsigned short f2bf(float f) {
    unsigned int u = __builtin_bit_cast(unsigned int, f);
    u += 0x7FFFu + ((u >> 16) & 1u);   // round-to-nearest-even
    return (unsigned short)(u >> 16);
}

// Build a v16bf WMMA fragment from two 16-byte chunks.
__device__ __forceinline__ v16bf ld_pair(const unsigned short* p0, const unsigned short* p1) {
    v8bf lo = *(const v8bf*)p0;
    v8bf hi = *(const v8bf*)p1;
    return __builtin_shufflevector(lo, hi, 0,1,2,3,4,5,6,7,8,9,10,11,12,13,14,15);
}

// ---------------------------------------------------------------- weights f32 -> bf16
__global__ void f32_to_bf16(const float* __restrict__ src, unsigned short* __restrict__ dst, int n) {
    int i = blockIdx.x * blockDim.x + threadIdx.x;
    if (i < n) dst[i] = f2bf(src[i]);
}

// ---------------------------------------------------------------- gather kernels (bf16 concat buffers)
__global__ void gather_char(const int* __restrict__ gid, const int* __restrict__ eid,
                            const int* __restrict__ aid, const int* __restrict__ hid,
                            const int* __restrict__ bid,
                            const float* __restrict__ Eg, const float* __restrict__ Ee,
                            const float* __restrict__ Ea, const float* __restrict__ Eh,
                            const float* __restrict__ Eb,
                            unsigned short* __restrict__ out) {   // [B,320]
    int row = blockIdx.x, t = threadIdx.x;
    int seg = t >> 6, off = t & 63;
    float v;
    if      (seg == 0) v = Eg[gid[row] * 64 + off];
    else if (seg == 1) v = Ee[eid[row] * 64 + off];
    else if (seg == 2) v = Ea[aid[row] * 64 + off];
    else if (seg == 3) v = Eh[hid[row] * 64 + off];
    else               v = Eb[bid[row] * 64 + off];
    out[(size_t)row * 320 + t] = f2bf(v);
}

__global__ void gather_outfit(const int* __restrict__ sid, const int* __restrict__ fid,
                              const int* __restrict__ tid, const int* __restrict__ bid,
                              const int* __restrict__ shid, const int* __restrict__ accid,
                              const float* __restrict__ Es, const float* __restrict__ Ef,
                              const float* __restrict__ Et, const float* __restrict__ Eb,
                              const float* __restrict__ Esh, const float* __restrict__ Eacc,
                              unsigned short* __restrict__ out) { // [B,384]
    int row = blockIdx.x, t = threadIdx.x;
    int seg = t / 64, off = t & 63;
    float v;
    if      (seg == 0) v = Es [sid [row] * 64 + off];
    else if (seg == 1) v = Ef [fid [row] * 64 + off];
    else if (seg == 2) v = Et [tid [row] * 64 + off];
    else if (seg == 3) v = Eb [bid [row] * 64 + off];
    else if (seg == 4) v = Esh[shid[row] * 64 + off];
    else {
        float s = 0.f;
        #pragma unroll
        for (int j = 0; j < 5; ++j) s += Eacc[accid[row * 5 + j] * 64 + off];
        v = s * 0.2f;
    }
    out[(size_t)row * 384 + t] = f2bf(v);
}

__global__ void gather_pose(const int* __restrict__ mid, const int* __restrict__ spid,
                            const int* __restrict__ hid, const int* __restrict__ toid,
                            const int* __restrict__ did,
                            const float* __restrict__ Em, const float* __restrict__ Esp,
                            const float* __restrict__ Eh, const float* __restrict__ Eto,
                            const float* __restrict__ Ed,
                            unsigned short* __restrict__ out) {   // [B,512]
    int row = blockIdx.x, t = threadIdx.x;
    float v;
    if      (t < 128) v = Em [mid [row] * 128 + t];
    else if (t < 256) v = Esp[spid[row] * 128 + (t - 128)];
    else if (t < 320) v = Eh [hid [row] * 64  + (t - 256)];
    else if (t < 384) v = Eto[toid[row] * 64  + (t - 320)];
    else              v = Ed [did [row] * 128 + (t - 384)];
    out[(size_t)row * 512 + t] = f2bf(v);
}

__global__ void gather_env(const int* __restrict__ lid, const int* __restrict__ bid,
                           const int* __restrict__ did, const int* __restrict__ aid,
                           const float* __restrict__ El, const float* __restrict__ Eb,
                           const float* __restrict__ Ed, const float* __restrict__ Ea,
                           unsigned short* __restrict__ out) {    // [B,256]
    int row = blockIdx.x, t = threadIdx.x;
    int seg = t >> 6, off = t & 63;
    float v;
    if      (seg == 0) v = El[lid[row] * 64 + off];
    else if (seg == 1) v = Eb[bid[row] * 64 + off];
    else if (seg == 2) v = Ed[did[row] * 64 + off];
    else               v = Ea[aid[row] * 64 + off];
    out[(size_t)row * 256 + t] = f2bf(v);
}

// ---------------------------------------------------------------- WMMA GEMM  out = A[B,K](bf16) x W[N,K]^T + bias
// 16-row strip per workgroup, 8 waves, TPW n-tiles per wave (N = TPW*128).
// EPI: 0 = ReLU->bf16   1 = ReLU+LayerNorm->bf16 (N==512)
//      2 = LayerNorm->bf16 (N==512)   3 = (1+sigmoid)->f32
template<int TPW, int EPI>
__global__ __launch_bounds__(256) void gemm16(
    const unsigned short* __restrict__ A, int lda,
    const unsigned short* __restrict__ W, int K,
    const float* __restrict__ bias,
    const float* __restrict__ gamma, const float* __restrict__ beta,
    unsigned short* __restrict__ outb, int ldob, int coloff,
    float* __restrict__ outf, int ldof)
{
    const int m0   = blockIdx.x * 16;
    const int wave = threadIdx.x >> 5;
    const int lane = threadIdx.x & 31;
    const int halfu = lane >> 4;
    const int l16  = lane & 15;

    v8f acc[TPW];
    #pragma unroll
    for (int t = 0; t < TPW; ++t) acc[t] = v8f{};

    const unsigned short* arow = A + (size_t)(m0 + l16) * lda + (halfu ? 8 : 0);

    for (int k = 0; k < K; k += 32) {
        v16bf af = ld_pair(arow + k, arow + k + 16);
        #pragma unroll
        for (int t = 0; t < TPW; ++t) {
            int n = ((wave + t * 8) << 4) + l16;
            const unsigned short* bp = W + (size_t)n * K + k + (halfu ? 8 : 0);
            v16bf bb = ld_pair(bp, bp + 16);
            acc[t] = __builtin_amdgcn_wmma_f32_16x16x32_bf16(
                false, af, false, bb, (short)0, acc[t], false, false);
        }
    }

    if constexpr (EPI == 0) {
        #pragma unroll
        for (int t = 0; t < TPW; ++t) {
            int nc = ((wave + t * 8) << 4) + l16;
            float bv = bias[nc];
            #pragma unroll
            for (int r = 0; r < 8; ++r) {
                float v = acc[t][r] + bv;
                v = v > 0.f ? v : 0.f;
                outb[(size_t)(m0 + r + 8 * halfu) * ldob + coloff + nc] = f2bf(v);
            }
        }
    } else if constexpr (EPI == 3) {
        #pragma unroll
        for (int t = 0; t < TPW; ++t) {
            int nc = ((wave + t * 8) << 4) + l16;
            float bv = bias[nc];
            #pragma unroll
            for (int r = 0; r < 8; ++r) {
                float v = acc[t][r] + bv;
                outf[(size_t)(m0 + r + 8 * halfu) * ldof + nc] = 1.0f + 1.0f / (1.0f + __expf(-v));
            }
        }
    } else {
        __shared__ float sm[16 * 512];
        __shared__ float red0[256];
        __shared__ float red1[256];
        __shared__ float mu[16];
        __shared__ float ri[16];
        #pragma unroll
        for (int t = 0; t < TPW; ++t) {
            int nc = ((wave + t * 8) << 4) + l16;
            float bv = bias[nc];
            #pragma unroll
            for (int r = 0; r < 8; ++r) {
                float v = acc[t][r] + bv;
                if (EPI == 1) v = v > 0.f ? v : 0.f;
                sm[(r + 8 * halfu) * 512 + nc] = v;
            }
        }
        __syncthreads();
        int row = threadIdx.x >> 4, c0 = threadIdx.x & 15;
        float s = 0.f, ss = 0.f;
        for (int c = c0; c < 512; c += 16) { float v = sm[row * 512 + c]; s += v; ss += v * v; }
        red0[threadIdx.x] = s; red1[threadIdx.x] = ss;
        __syncthreads();
        if (threadIdx.x < 16) {
            float ts = 0.f, tss = 0.f;
            for (int j = 0; j < 16; ++j) { ts += red0[threadIdx.x * 16 + j]; tss += red1[threadIdx.x * 16 + j]; }
            float m = ts * (1.0f / 512.0f);
            float var = tss * (1.0f / 512.0f) - m * m;
            mu[threadIdx.x] = m;
            ri[threadIdx.x] = rsqrtf(var + LNEPS);
        }
        __syncthreads();
        float m = mu[row], rr = ri[row];
        for (int c = c0; c < 512; c += 16) {
            float v = (sm[row * 512 + c] - m) * rr * gamma[c] + beta[c];
            outb[(size_t)(m0 + row) * ldob + coloff + c] = f2bf(v);
        }
    }
}

// ---------------------------------------------------------------- fused 3-layer scene MLP (h1/h2 kept in LDS)
__global__ __launch_bounds__(256) void scene_fused(
    const unsigned short* __restrict__ SC,  // [B,2048] bf16
    const unsigned short* __restrict__ W1, const float* __restrict__ b1,   // [1536,2048]
    const unsigned short* __restrict__ W2, const float* __restrict__ b2,   // [1024,1536]
    const unsigned short* __restrict__ W3, const float* __restrict__ b3,   // [512,1024]
    const float* __restrict__ gamma, const float* __restrict__ beta,
    const float* __restrict__ gate,         // [B,512] holds 1+sigmoid
    float* __restrict__ out)                // [B,512]
{
    __shared__ __align__(16) unsigned char lds[16 * 1536 * 2 + 16 * 1024 * 2 + 2048 + 128];
    unsigned short* h1s = (unsigned short*)lds;                         // 48 KB
    unsigned short* h2s = (unsigned short*)(lds + 16 * 1536 * 2);       // 32 KB
    float* smf  = (float*)lds;                                          // aliases h1s (safe: h1s dead)
    float* red0 = (float*)(lds + 16 * 1536 * 2 + 16 * 1024 * 2);
    float* red1 = red0 + 256;
    float* mu   = red1 + 256;
    float* ri   = mu + 16;

    const int m0   = blockIdx.x * 16;
    const int wave = threadIdx.x >> 5;
    const int lane = threadIdx.x & 31;
    const int halfu = lane >> 4;
    const int l16  = lane & 15;

    // ---------- layer 1: [16,2048] x [1536,2048]^T, ReLU -> h1s
    {
        v8f acc[12];
        #pragma unroll
        for (int t = 0; t < 12; ++t) acc[t] = v8f{};
        const unsigned short* arow = SC + (size_t)(m0 + l16) * 2048 + (halfu ? 8 : 0);
        for (int k = 0; k < 2048; k += 32) {
            v16bf af = ld_pair(arow + k, arow + k + 16);
            #pragma unroll
            for (int t = 0; t < 12; ++t) {
                int n = ((wave + t * 8) << 4) + l16;
                const unsigned short* bp = W1 + (size_t)n * 2048 + k + (halfu ? 8 : 0);
                acc[t] = __builtin_amdgcn_wmma_f32_16x16x32_bf16(
                    false, af, false, ld_pair(bp, bp + 16), (short)0, acc[t], false, false);
            }
        }
        #pragma unroll
        for (int t = 0; t < 12; ++t) {
            int nc = ((wave + t * 8) << 4) + l16;
            float bv = b1[nc];
            #pragma unroll
            for (int r = 0; r < 8; ++r) {
                float v = acc[t][r] + bv;
                v = v > 0.f ? v : 0.f;
                h1s[(r + 8 * halfu) * 1536 + nc] = f2bf(v);
            }
        }
    }
    __syncthreads();

    // ---------- layer 2: [16,1536] x [1024,1536]^T, ReLU -> h2s
    {
        v8f acc[8];
        #pragma unroll
        for (int t = 0; t < 8; ++t) acc[t] = v8f{};
        const unsigned short* arow = h1s + l16 * 1536 + (halfu ? 8 : 0);
        for (int k = 0; k < 1536; k += 32) {
            v16bf af = ld_pair(arow + k, arow + k + 16);
            #pragma unroll
            for (int t = 0; t < 8; ++t) {
                int n = ((wave + t * 8) << 4) + l16;
                const unsigned short* bp = W2 + (size_t)n * 1536 + k + (halfu ? 8 : 0);
                acc[t] = __builtin_amdgcn_wmma_f32_16x16x32_bf16(
                    false, af, false, ld_pair(bp, bp + 16), (short)0, acc[t], false, false);
            }
        }
        __syncthreads();   // all h1s reads done before any h2s consumer; also keeps phases ordered
        #pragma unroll
        for (int t = 0; t < 8; ++t) {
            int nc = ((wave + t * 8) << 4) + l16;
            float bv = b2[nc];
            #pragma unroll
            for (int r = 0; r < 8; ++r) {
                float v = acc[t][r] + bv;
                v = v > 0.f ? v : 0.f;
                h2s[(r + 8 * halfu) * 1024 + nc] = f2bf(v);
            }
        }
    }
    __syncthreads();

    // ---------- layer 3: [16,1024] x [512,1024]^T -> smf, then LN * gate -> out
    {
        v8f acc[4];
        #pragma unroll
        for (int t = 0; t < 4; ++t) acc[t] = v8f{};
        const unsigned short* arow = h2s + l16 * 1024 + (halfu ? 8 : 0);
        for (int k = 0; k < 1024; k += 32) {
            v16bf af = ld_pair(arow + k, arow + k + 16);
            #pragma unroll
            for (int t = 0; t < 4; ++t) {
                int n = ((wave + t * 8) << 4) + l16;
                const unsigned short* bp = W3 + (size_t)n * 1024 + k + (halfu ? 8 : 0);
                acc[t] = __builtin_amdgcn_wmma_f32_16x16x32_bf16(
                    false, af, false, ld_pair(bp, bp + 16), (short)0, acc[t], false, false);
            }
        }
        #pragma unroll
        for (int t = 0; t < 4; ++t) {
            int nc = ((wave + t * 8) << 4) + l16;
            float bv = b3[nc];
            #pragma unroll
            for (int r = 0; r < 8; ++r)
                smf[(r + 8 * halfu) * 512 + nc] = acc[t][r] + bv;
        }
    }
    __syncthreads();

    int row = threadIdx.x >> 4, c0 = threadIdx.x & 15;
    float s = 0.f, ss = 0.f;
    for (int c = c0; c < 512; c += 16) { float v = smf[row * 512 + c]; s += v; ss += v * v; }
    red0[threadIdx.x] = s; red1[threadIdx.x] = ss;
    __syncthreads();
    if (threadIdx.x < 16) {
        float ts = 0.f, tss = 0.f;
        for (int j = 0; j < 16; ++j) { ts += red0[threadIdx.x * 16 + j]; tss += red1[threadIdx.x * 16 + j]; }
        float m = ts * (1.0f / 512.0f);
        float var = tss * (1.0f / 512.0f) - m * m;
        mu[threadIdx.x] = m;
        ri[threadIdx.x] = rsqrtf(var + LNEPS);
    }
    __syncthreads();
    float m = mu[row], rr = ri[row];
    for (int c = c0; c < 512; c += 16) {
        float v = (smf[row * 512 + c] - m) * rr * gamma[c] + beta[c];
        out[(size_t)(m0 + row) * 512 + c] = v * gate[(size_t)(m0 + row) * 512 + c];
    }
}

// ---------------------------------------------------------------- host launcher
extern "C" void kernel_launch(void* const* d_in, const int* in_sizes, int n_in,
                              void* d_out, int out_size, void* d_ws, size_t ws_size,
                              hipStream_t stream) {
    (void)in_sizes; (void)n_in; (void)out_size; (void)ws_size;

    // --- indices ---
    const int* gender = (const int*)d_in[0];  const int* eth    = (const int*)d_in[1];
    const int* age    = (const int*)d_in[2];  const int* height = (const int*)d_in[3];
    const int* body   = (const int*)d_in[4];  const int* style  = (const int*)d_in[5];
    const int* form   = (const int*)d_in[6];  const int* top    = (const int*)d_in[7];
    const int* bottom = (const int*)d_in[8];  const int* shoe   = (const int*)d_in[9];
    const int* accids = (const int*)d_in[10]; const int* motion = (const int*)d_in[11];
    const int* speed  = (const int*)d_in[12]; const int* head   = (const int*)d_in[13];
    const int* torso  = (const int*)d_in[14]; const int* dir    = (const int*)d_in[15];
    const int* light  = (const int*)d_in[16]; const int* bg     = (const int*)d_in[17];
    const int* dist   = (const int*)d_in[18]; const int* angle  = (const int*)d_in[19];
    // --- embedding tables ---
    const float* E_gender = (const float*)d_in[20]; const float* E_eth  = (const float*)d_in[21];
    const float* E_age    = (const float*)d_in[22]; const float* E_h    = (const float*)d_in[23];
    const float* E_body   = (const float*)d_in[24]; const float* E_sty  = (const float*)d_in[25];
    const float* E_form   = (const float*)d_in[26]; const float* E_top  = (const float*)d_in[27];
    const float* E_bot    = (const float*)d_in[28]; const float* E_shoe = (const float*)d_in[29];
    const float* E_acc    = (const float*)d_in[30]; const float* E_mot  = (const float*)d_in[31];
    const float* E_spd    = (const float*)d_in[32]; const float* E_head = (const float*)d_in[33];
    const float* E_tor    = (const float*)d_in[34]; const float* E_dir  = (const float*)d_in[35];
    const float* E_light  = (const float*)d_in[36]; const float* E_bg   = (const float*)d_in[37];
    const float* E_dist   = (const float*)d_in[38]; const float* E_ang  = (const float*)d_in[39];
    // --- params ---
    const float* char_W  = (const float*)d_in[40]; const float* char_b  = (const float*)d_in[41];
    const float* char_g  = (const float*)d_in[42]; const float* char_be = (const float*)d_in[43];
    const float* outf_W  = (const float*)d_in[44]; const float* outf_b  = (const float*)d_in[45];
    const float* outf_g  = (const float*)d_in[46]; const float* outf_be = (const float*)d_in[47];
    const float* pose_W1 = (const float*)d_in[48]; const float* pose_b1 = (const float*)d_in[49];
    const float* pose_W2 = (const float*)d_in[50]; const float* pose_b2 = (const float*)d_in[51];
    const float* pose_g  = (const float*)d_in[52]; const float* pose_be = (const float*)d_in[53];
    const float* env_W   = (const float*)d_in[54]; const float* env_b   = (const float*)d_in[55];
    const float* env_g   = (const float*)d_in[56]; const float* env_be  = (const float*)d_in[57];
    const float* sc_W1   = (const float*)d_in[58]; const float* sc_b1   = (const float*)d_in[59];
    const float* sc_W2   = (const float*)d_in[60]; const float* sc_b2   = (const float*)d_in[61];
    const float* sc_W3   = (const float*)d_in[62]; const float* sc_b3   = (const float*)d_in[63];
    const float* sc_g    = (const float*)d_in[64]; const float* sc_be   = (const float*)d_in[65];
    const float* gate_W  = (const float*)d_in[66]; const float* gate_b  = (const float*)d_in[67];

    // --- workspace carving ---
    char* ws = (char*)d_ws;
    size_t off = 0;
    auto carve = [&](size_t bytes) -> char* {
        char* p = ws + off;
        off = (off + bytes + 255) & ~(size_t)255;
        return p;
    };
    unsigned short* wb_char = (unsigned short*)carve((size_t)512 * 320  * 2);
    unsigned short* wb_outf = (unsigned short*)carve((size_t)512 * 384  * 2);
    unsigned short* wb_p1   = (unsigned short*)carve((size_t)1024 * 512 * 2);
    unsigned short* wb_p2   = (unsigned short*)carve((size_t)512 * 1024 * 2);
    unsigned short* wb_env  = (unsigned short*)carve((size_t)512 * 256  * 2);
    unsigned short* wb_gate = (unsigned short*)carve((size_t)512 * 512  * 2);
    unsigned short* wb_s1   = (unsigned short*)carve((size_t)1536 * 2048 * 2);
    unsigned short* wb_s2   = (unsigned short*)carve((size_t)1024 * 1536 * 2);
    unsigned short* wb_s3   = (unsigned short*)carve((size_t)512 * 1024 * 2);
    unsigned short* scene_c = (unsigned short*)carve((size_t)NROWS * 2048 * 2);
    unsigned short* char_c  = (unsigned short*)carve((size_t)NROWS * 320 * 2);
    unsigned short* outf_c  = (unsigned short*)carve((size_t)NROWS * 384 * 2);
    unsigned short* pose_c  = (unsigned short*)carve((size_t)NROWS * 512 * 2);
    unsigned short* env_c   = (unsigned short*)carve((size_t)NROWS * 256 * 2);
    unsigned short* pose_h  = (unsigned short*)carve((size_t)NROWS * 1024 * 2);
    float* gatebuf = (float*)pose_h;   // alias: pose_h dead before gate GEMM writes (same byte size)

    auto conv = [&](const float* s, unsigned short* d, int n) {
        f32_to_bf16<<<(n + 255) / 256, 256, 0, stream>>>(s, d, n);
    };
    conv(char_W,  wb_char, 512 * 320);
    conv(outf_W,  wb_outf, 512 * 384);
    conv(pose_W1, wb_p1,   1024 * 512);
    conv(pose_W2, wb_p2,   512 * 1024);
    conv(env_W,   wb_env,  512 * 256);
    conv(gate_W,  wb_gate, 512 * 512);
    conv(sc_W1,   wb_s1,   1536 * 2048);
    conv(sc_W2,   wb_s2,   1024 * 1536);
    conv(sc_W3,   wb_s3,   512 * 1024);

    gather_char  <<<NROWS, 320, 0, stream>>>(gender, eth, age, height, body,
                                             E_gender, E_eth, E_age, E_h, E_body, char_c);
    gather_outfit<<<NROWS, 384, 0, stream>>>(style, form, top, bottom, shoe, accids,
                                             E_sty, E_form, E_top, E_bot, E_shoe, E_acc, outf_c);
    gather_pose  <<<NROWS, 512, 0, stream>>>(motion, speed, head, torso, dir,
                                             E_mot, E_spd, E_head, E_tor, E_dir, pose_c);
    gather_env   <<<NROWS, 256, 0, stream>>>(light, bg, dist, angle,
                                             E_light, E_bg, E_dist, E_ang, env_c);

    const int GB = NROWS / 16;   // 2048 workgroups, 16-row strips
    // branch GEMMs -> bf16 scene_c (concat for free)
    gemm16<4, 1><<<GB, 256, 0, stream>>>(char_c, 320, wb_char, 320, char_b, char_g, char_be,
                                         scene_c, 2048, 0,    nullptr, 0);
    gemm16<4, 1><<<GB, 256, 0, stream>>>(outf_c, 384, wb_outf, 384, outf_b, outf_g, outf_be,
                                         scene_c, 2048, 512,  nullptr, 0);
    gemm16<4, 1><<<GB, 256, 0, stream>>>(env_c,  256, wb_env,  256, env_b, env_g, env_be,
                                         scene_c, 2048, 1536, nullptr, 0);
    // pose 2-layer MLP
    gemm16<8, 0><<<GB, 256, 0, stream>>>(pose_c, 512, wb_p1, 512, pose_b1, nullptr, nullptr,
                                         pose_h, 1024, 0, nullptr, 0);
    gemm16<4, 2><<<GB, 256, 0, stream>>>(pose_h, 1024, wb_p2, 1024, pose_b2, pose_g, pose_be,
                                         scene_c, 2048, 1024, nullptr, 0);
    // gate = 1 + sigmoid(pose_e . gate_W^T + b) ; pose_e read back from scene_c cols [1024,1536)
    gemm16<4, 3><<<GB, 256, 0, stream>>>(scene_c + 1024, 2048, wb_gate, 512, gate_b, nullptr, nullptr,
                                         nullptr, 0, 0, gatebuf, 512);
    // fused scene MLP + LN + gate multiply -> d_out (f32)
    scene_fused<<<GB, 256, 0, stream>>>(scene_c, wb_s1, sc_b1, wb_s2, sc_b2, wb_s3, sc_b3,
                                        sc_g, sc_be, gatebuf, (float*)d_out);
}
